// PositizedLinear_49014166782407
// MI455X (gfx1250) — compile-verified, matrix operand order checked by reading, not run on hardware
//
#include <hip/hip_runtime.h>

// ---------------------------------------------------------------------------
// PositizedLinear for MI455X (gfx1250, wave32, WMMA):
//   y = posit8_1(x) @ posit8_1(W)^T + posit8_1(b)
// posit(8,1) values are exactly representable in FP16 (<=4 fraction bits,
// |v| in [2^-12, 2^12]) -> quantize to _Float16, GEMM on v_wmma_f32_16x16x32_f16
// with f32 accumulation. GEMM stages tiles in LDS via async global->LDS
// copies (ASYNCcnt double-buffering), 4x4 register tiles per wave, and a
// software double-buffered B fragment to hide DS latency between WMMA quads.
// ---------------------------------------------------------------------------

typedef __attribute__((ext_vector_type(16))) _Float16 v16h;
typedef __attribute__((ext_vector_type(4)))  _Float16 v4h;
typedef __attribute__((ext_vector_type(8)))  float    v8f;
typedef __attribute__((ext_vector_type(4)))  unsigned u32x4;

#define IN_F   2048
#define OUT_F  2048
#define NROWS  8192

// --------- gfx1250 async global->LDS plumbing (guarded, with fallback) -----
#if __has_builtin(__builtin_amdgcn_global_load_async_to_lds_b128)
#define HAVE_ASYNC_LDS 1
typedef int v4i_vs __attribute__((vector_size(16)));
typedef __attribute__((address_space(1))) v4i_vs* gvec_ptr;
typedef __attribute__((address_space(3))) v4i_vs* lvec_ptr;
#define ASYNC_CP_B128(l, g)                                                   \
    __builtin_amdgcn_global_load_async_to_lds_b128((gvec_ptr)(g),             \
                                                   (lvec_ptr)(l), 0, 0)
#if __has_builtin(__builtin_amdgcn_s_wait_asynccnt)
#define WAIT_ASYNCCNT(n) __builtin_amdgcn_s_wait_asynccnt(n)
#else
#define WAIT_ASYNCCNT(n) asm volatile("s_wait_asynccnt %0" ::"n"(n) : "memory")
#endif
#else
#define HAVE_ASYNC_LDS 0
#define ASYNC_CP_B128(l, g) (*(u32x4*)(l) = *(const u32x4*)(g))
#define WAIT_ASYNCCNT(n) do { } while (0)
#endif

// ---------------- posit(8,1) round-to-nearest quantizer --------------------
__device__ __forceinline__ float posit8_1(float x) {
    if (x == 0.0f) return 0.0f;
    const float maxpos = 4096.0f;          // 2^12
    const float minpos = 1.0f / 4096.0f;   // 2^-12
    float ax = __builtin_fabsf(x);
    ax = fminf(fmaxf(ax, minpos), maxpos);
    int s = (int)((__float_as_uint(ax) >> 23) & 0xFFu) - 127; // exact floor(log2)
    int k = s >> 1;                         // floor(s/2)
    int rlen = (k >= 0) ? (k + 2) : (1 - k);
    int fbits = 6 - rlen;
    fbits = fbits < 0 ? 0 : fbits;
    float m = ax * __uint_as_float((unsigned)(127 - s) << 23);      // [1,2)
    float fscale  = __uint_as_float((unsigned)(127 + fbits) << 23); // 2^fbits
    float ifscale = __uint_as_float((unsigned)(127 - fbits) << 23); // 2^-fbits
    float mq = rintf(m * fscale) * ifscale; // ties-to-even
    if (mq >= 2.0f) { s += 1; mq = 1.0f; }  // mantissa carry
    float y = mq * __uint_as_float((unsigned)(s + 127) << 23);
    y = fminf(y, maxpos);
    return copysignf(y, x);
}

__global__ __launch_bounds__(256) void posit_quant_kernel(
    const float* __restrict__ in, _Float16* __restrict__ out, int n4)
{
    int i      = blockIdx.x * blockDim.x + threadIdx.x;
    int stride = gridDim.x * blockDim.x;
    const float4* in4 = (const float4*)in;
    v4h* out4 = (v4h*)out;
    for (; i < n4; i += stride) {
        float4 v = in4[i];
        v4h o;
        o[0] = (_Float16)posit8_1(v.x);
        o[1] = (_Float16)posit8_1(v.y);
        o[2] = (_Float16)posit8_1(v.z);
        o[3] = (_Float16)posit8_1(v.w);
        out4[i] = o;
    }
}

// ---------------- WMMA GEMM with async-LDS double buffering ----------------
// Workgroup: 256 threads = 8 waves as 4(M) x 2(N); per-wave 4x4 tiles of
// 16x16 -> 64x64 per wave, 256(M) x 128(N) per workgroup. K staged in steps
// of 32 into LDS double buffers (padded 80B row stride -> conflict-free
// ds_load_b128 fragment reads). Each thread issues 6 async b128 copies per
// K-step; s_wait_asynccnt 6 overlaps the next slab's DMA with WMMA.
#define BLK_M 256
#define BLK_N 128
#define TK    32
#define ROWH  40                        // LDS row stride in halfs (80 B, padded)
#define A_HALFS (BLK_M * ROWH)          // 10240 halfs per buffer
#define B_HALFS (BLK_N * ROWH)          //  5120 halfs per buffer
#define B_BASE  (2 * A_HALFS)           // B region starts after both A buffers

__global__ __launch_bounds__(256) void posit_gemm_wmma(
    const _Float16* __restrict__ xq, const _Float16* __restrict__ wq,
    const _Float16* __restrict__ bq, float* __restrict__ out)
{
    __shared__ _Float16 smem[2 * A_HALFS + 2 * B_HALFS];   // 61440 bytes

    const int t    = threadIdx.x;
    const int lane = t & 31;
    const int wave = t >> 5;
    const int wm   = wave >> 1;                 // 0..3 (M)
    const int wn   = wave & 1;                  // 0..1 (N)
    const int blockM = blockIdx.y * BLK_M;
    const int blockN = blockIdx.x * BLK_N;

    // NOTE: no pointer arrays into __shared__ (lld rejects addrspacecast in
    // static initializers) -- buffer addresses are computed arithmetically.

    // Stage one K-slab (TK halfs deep) into LDS buffer `buf`.
    // A: 256 rows x 4 chunks, B: 128 rows x 4 chunks; chunk = 16B (8 halfs).
    auto stage = [&](int buf, int k0) {
        #pragma unroll
        for (int j = 0; j < 4; ++j) {
            int id = t + j * 256;               // 0..1023
            int r = id >> 2, c = id & 3;
            ASYNC_CP_B128(smem + buf * A_HALFS + r * ROWH + c * 8,
                          xq + (size_t)(blockM + r) * IN_F + k0 + c * 8);
        }
        #pragma unroll
        for (int j = 0; j < 2; ++j) {
            int id = t + j * 256;               // 0..511
            int r = id >> 2, c = id & 3;
            ASYNC_CP_B128(smem + B_BASE + buf * B_HALFS + r * ROWH + c * 8,
                          wq + (size_t)(blockN + r) * IN_F + k0 + c * 8);
        }
    };

    v8f acc[4][4] = {};
    // Fragment layout (16-bit A-layout, both operands K-contiguous):
    // lane&15 = tile row; lanes>=16 carry K 8..15 & 24..31 (coff = 8 halfs).
    const int rowA0 = wm * 64 + (lane & 15);
    const int rowB0 = wn * 64 + (lane & 15);
    const int coff  = (lane >> 4) << 3;

    stage(0, 0);

    union Frag { v16h h; u32x4 q[2]; };

    int buf = 0;
    #pragma unroll 1
    for (int k0 = 0; k0 < IN_F; k0 += TK, buf ^= 1) {
        if (k0 + TK < IN_F) { stage(buf ^ 1, k0 + TK); WAIT_ASYNCCNT(6); }
        else                { WAIT_ASYNCCNT(0); }
        __syncthreads();    // current buffer fully visible to all waves

        auto loadB = [&](Frag& f, int ni) {
            const _Float16* l =
                smem + B_BASE + buf * B_HALFS + (rowB0 + ni * 16) * ROWH + coff;
            f.q[0] = *(const u32x4*)(l);
            f.q[1] = *(const u32x4*)(l + 16);
        };

        Frag a[4];
        #pragma unroll
        for (int mi = 0; mi < 4; ++mi) {
            const _Float16* l =
                smem + buf * A_HALFS + (rowA0 + mi * 16) * ROWH + coff;
            a[mi].q[0] = *(const u32x4*)(l);
            a[mi].q[1] = *(const u32x4*)(l + 16);
        }

        // Software double-buffered B fragment: issue B[ni+1] loads before the
        // WMMA quad for B[ni] so ds latency hides under WMMA issue.
        Frag b[2];
        loadB(b[0], 0);
        #pragma unroll
        for (int ni = 0; ni < 4; ++ni) {
            if (ni < 3) loadB(b[(ni + 1) & 1], ni + 1);
            #pragma unroll
            for (int mi = 0; mi < 4; ++mi)
                acc[mi][ni] = __builtin_amdgcn_wmma_f32_16x16x32_f16(
                    false, a[mi].h, false, b[ni & 1].h, (short)0, acc[mi][ni],
                    false, false);
        }
        __syncthreads();    // all reads done before next stage overwrites
    }

    // Epilogue: D layout = 8 VGPRs f32; lane&15 = column, lanes>=16 -> rows+8.
    const int colT   = lane & 15;
    const int rowOff = (lane >> 4) << 3;
    #pragma unroll
    for (int ni = 0; ni < 4; ++ni) {
        const int col = blockN + wn * 64 + ni * 16 + colT;
        const float bv = (float)bq[col];
        #pragma unroll
        for (int mi = 0; mi < 4; ++mi) {
            float* o = out + (size_t)(blockM + wm * 64 + mi * 16 + rowOff) * OUT_F + col;
            #pragma unroll
            for (int r = 0; r < 8; ++r)
                o[(size_t)r * OUT_F] = acc[mi][ni][r] + bv;
        }
    }
}

extern "C" void kernel_launch(void* const* d_in, const int* in_sizes, int n_in,
                              void* d_out, int out_size, void* d_ws, size_t ws_size,
                              hipStream_t stream) {
    const float* x = (const float*)d_in[0];
    const float* w = (const float*)d_in[1];
    const float* b = (const float*)d_in[2];
    float* out = (float*)d_out;

    // Workspace layout: xq (32MB) | wq (8MB) | bq (4KB) as _Float16
    _Float16* xq = (_Float16*)d_ws;
    _Float16* wq = xq + (size_t)NROWS * IN_F;
    _Float16* bq = wq + (size_t)OUT_F * IN_F;

    posit_quant_kernel<<<4096, 256, 0, stream>>>(x, xq, (NROWS * IN_F) / 4);
    posit_quant_kernel<<<4096, 256, 0, stream>>>(w, wq, (OUT_F * IN_F) / 4);
    posit_quant_kernel<<<2,    256, 0, stream>>>(b, bq, OUT_F / 4);

    dim3 grid(OUT_F / BLK_N, NROWS / BLK_M);   // (16, 32)
    posit_gemm_wmma<<<grid, 256, 0, stream>>>(xq, wq, bq, out);
}
